// SparseMultiModalEncoderPaint_80049600462904
// MI455X (gfx1250) — compile-verified, matrix operand order checked by reading, not run on hardware
//
#include <hip/hip_runtime.h>
#include <hip/hip_bf16.h>
#include <stdint.h>

// ---------------------------------------------------------------------------
// MI455X (gfx1250) sparse multi-modal encoder.
//
// Roofline: ~140 GFLOP total; dominated by six K=27 gather-GEMM convs whose
// activations (max 80000x259) all fit in the 192 MB L2 -> the 27-way random
// row gather is L2-bound. All intermediates are bf16 (half the gather bytes);
// GEMMs run on v_wmma_f32_16x16x32_bf16 (wave32, f32 accumulate).
//
// CDNA5 paths:
//   * v_wmma_f32_16x16x32_bf16, fully unrolled via compile-time Cin/Cout.
//   * GLOBAL_LOAD_ASYNC_TO_LDS_B128 + s_wait_asynccnt, double-buffered:
//     stage k+1 is issued before waiting on stage k (async loads complete
//     in order per wave), overlapping the L2 gather with WMMA.
//   * Two 16-row tiles per workgroup share each B fragment (weight reuse).
// Weights pre-transposed to [K][Cout][Cin_pad] bf16 so both WMMA fragments
// are two 16-byte vector loads matching the 16-bit A-matrix lane layout.
// ---------------------------------------------------------------------------

typedef __attribute__((ext_vector_type(16))) __bf16 v16bf;
typedef __attribute__((ext_vector_type(8)))  float  v8f;

#define TILE_M 16

__device__ __forceinline__ unsigned short f2bf(float f) {
  unsigned int u = __float_as_uint(f);
  u += 0x7FFFu + ((u >> 16) & 1u);  // round-to-nearest-even
  return (unsigned short)(u >> 16);
}
__device__ __forceinline__ float bf2f(unsigned short h) {
  return __uint_as_float(((unsigned int)h) << 16);
}

// async 16B copy: global -> LDS, tracked by ASYNCcnt.
__device__ __forceinline__ void async_copy_b128(unsigned lds_off, const void* gaddr) {
  asm volatile("global_load_async_to_lds_b128 %0, %1, off"
               :: "v"(lds_off), "v"(gaddr)
               : "memory");
}
template <int N>
__device__ __forceinline__ void wait_async() {
  asm volatile("s_wait_asynccnt %0" :: "n"(N) : "memory");
}

// ---------------------------------------------------------------------------
// Prep kernels
// ---------------------------------------------------------------------------
__global__ void wt_prep(const float* __restrict__ W, unsigned short* __restrict__ Wt,
                        int Kk, int Cin, int Cout, int Cin_pad) {
  long total = (long)Kk * Cout * Cin_pad;
  for (long e = (long)blockIdx.x * blockDim.x + threadIdx.x; e < total;
       e += (long)gridDim.x * blockDim.x) {
    int ci = (int)(e % Cin_pad);
    long t = e / Cin_pad;
    int co = (int)(t % Cout);
    int k  = (int)(t / Cout);
    float v = (ci < Cin) ? W[((long)k * Cin + ci) * Cout + co] : 0.f;
    Wt[e] = f2bf(v);
  }
}

__global__ void zero_u16(unsigned short* __restrict__ p, int n) {
  int i = blockIdx.x * blockDim.x + threadIdx.x;
  if (i < n) p[i] = 0;
}

__global__ void cvt_bf16(const float* __restrict__ x, unsigned short* __restrict__ y, long n) {
  for (long e = (long)blockIdx.x * blockDim.x + threadIdx.x; e < n;
       e += (long)gridDim.x * blockDim.x)
    y[e] = f2bf(x[e]);
}

__global__ void gather_cvt(const float* __restrict__ src, const int* __restrict__ idx,
                           unsigned short* __restrict__ dst, int n, int C) {
  long total = (long)n * C;
  for (long e = (long)blockIdx.x * blockDim.x + threadIdx.x; e < total;
       e += (long)gridDim.x * blockDim.x) {
    long r = e / C;
    int  c = (int)(e - r * C);
    int id = idx[r];
    dst[e] = (id >= 0) ? f2bf(src[(long)id * C + c]) : (unsigned short)0;
  }
}

__global__ void ew_mul_bf16(unsigned short* __restrict__ a,
                            const unsigned short* __restrict__ b, long n) {
  for (long e = (long)blockIdx.x * blockDim.x + threadIdx.x; e < n;
       e += (long)gridDim.x * blockDim.x)
    a[e] = f2bf(bf2f(a[e]) * bf2f(b[e]));
}

__global__ void concat_mix(const unsigned short* __restrict__ y3d,
                           const unsigned short* __restrict__ x2d,
                           const unsigned short* __restrict__ cr2d,
                           unsigned short* __restrict__ mix, int n) {
  long total = (long)n * 80;
  for (long e = (long)blockIdx.x * blockDim.x + threadIdx.x; e < total;
       e += (long)gridDim.x * blockDim.x) {
    long r = e / 80;
    int  c = (int)(e - r * 80);
    unsigned short v;
    if (c < 16) v = y3d[r * 16 + c];
    else {
      int cc = c - 16;
      v = f2bf(bf2f(x2d[r * 64 + cc]) + bf2f(cr2d[r * 64 + cc]));
    }
    mix[e] = v;
  }
}

// ---------------------------------------------------------------------------
// Fused gather-GEMM, compile-time (CIN, COUT, ASYNC), ROWS=2 16-row tiles
// per workgroup.  out[r] = epi( sum_k feats[nbr[r,k]] @ Wt[k] ).
// ---------------------------------------------------------------------------
template <int CIN, int COUT, bool ASYNC>
__global__ void __launch_bounds__(256)
conv_wmma(const unsigned short* __restrict__ feats,   // [n_in][CIN] bf16
          const int* __restrict__ nbr,                // [n_out][Kk] or null
          const unsigned short* __restrict__ Wt,      // [Kk][COUT][CIN_PAD] bf16
          const float* __restrict__ gamma,            // [COUT] or null
          const float* __restrict__ beta,             // [COUT] or null
          const unsigned short* __restrict__ resid,   // [n_out][COUT] bf16 or null
          const unsigned short* __restrict__ zrow,    // >=CIN bf16 zeros (global)
          void* __restrict__ out,                     // [n_out][COUT]
          int n_out, int Kk, int relu_flag, int out_f32) {
  constexpr int ROWS    = 2;                          // 16-row tiles / workgroup
  constexpr int MROWS   = ROWS * TILE_M;              // 32 staged rows
  constexpr int CIN_PAD = (CIN + 31) & ~31;
  constexpr int CHUNKS  = CIN_PAD / 32;               // WMMA K-steps per k
  constexpr int NTILES  = (COUT + 15) / 16;
  constexpr int NWAVES  = NTILES < 8 ? NTILES : 8;
  constexpr int TPW     = (NTILES + NWAVES - 1) / NWAVES;
  constexpr int NTHREADS = 32 * NWAVES;
  constexpr int BUFELEMS = MROWS * CIN_PAD;           // bf16 per LDS buffer
  // async staging: 16B chunks, uniform SI async issues per wave per stage
  constexpr int CPR   = CIN / 8;                      // chunks per row (ASYNC)
  constexpr int TOTAL = MROWS * CPR;
  constexpr int CPWV  = (TOTAL + NWAVES - 1) / NWAVES;
  constexpr int SI    = (CPWV + 31) / 32;

  __shared__ int sIdx[MROWS * 28];                    // all-k gather indices
  extern __shared__ uint4 lds_raw[];
  unsigned short* ldsA = (unsigned short*)lds_raw;    // 1 or 2 buffers

  const int tid  = threadIdx.x;
  const int lane = tid & 31;
  const int wave = tid >> 5;
  const int row0 = blockIdx.x * MROWS;

  // Stage all gather indices up front (one barrier for all k).
  for (int e = tid; e < MROWS * Kk; e += NTHREADS) {
    int k = e / MROWS, r = e - k * MROWS;
    int rr = row0 + r;
    sIdx[k * MROWS + r] = (rr < n_out) ? (nbr ? nbr[(long)rr * Kk + k] : rr) : -1;
  }
  if constexpr (ASYNC && (CIN_PAD > CIN)) {           // zero pad cols once (both bufs)
    constexpr int PADQ = (CIN_PAD - CIN) / 8;
    for (int e = tid; e < 2 * MROWS * PADQ; e += NTHREADS) {
      int r = e / PADQ, q = e - r * PADQ;             // r spans both buffers
      *(uint4*)(ldsA + r * CIN_PAD + CIN + q * 8) = make_uint4(0u, 0u, 0u, 0u);
    }
  }
  __syncthreads();

  v8f zero = {0.f, 0.f, 0.f, 0.f, 0.f, 0.f, 0.f, 0.f};
  v8f acc[TPW][ROWS];
#pragma unroll
  for (int tt = 0; tt < TPW; ++tt)
#pragma unroll
    for (int g = 0; g < ROWS; ++g) acc[tt][g] = zero;

  const int mrow = lane & 15;
  const int klo  = (lane >> 4) << 3;                  // 0 / 8 bf16 lane K-split

  // issue one stage: exactly SI async instructions per wave (uniform).
  auto stage = [&](int k, int buf) {
    unsigned short* dbuf = ldsA + buf * BUFELEMS;
#pragma unroll
    for (int i = 0; i < SI; ++i) {
      int e = wave * CPWV + i * 32 + lane;
      if (e >= TOTAL) e -= TOTAL;                     // wrap: duplicate writes ok
      int r = e / CPR, q = e - r * CPR;
      int id = sIdx[k * MROWS + r];
      const unsigned short* src =
          (id >= 0) ? (feats + (long)id * CIN + q * 8) : zrow;  // zeros if missing
      async_copy_b128((unsigned)(uintptr_t)(dbuf + r * CIN_PAD + q * 8),
                      (const void*)src);
    }
  };

  if constexpr (ASYNC) stage(0, 0);

  for (int k = 0; k < Kk; ++k) {
    const int buf = ASYNC ? (k & 1) : 0;
    if constexpr (ASYNC) {
      if (k + 1 < Kk) { stage(k + 1, (k + 1) & 1); wait_async<SI>(); }
      else            { wait_async<0>(); }
    } else {
      constexpr int LG = NTHREADS / MROWS;            // threads per staged row
      const int lrow = tid / LG, lc0 = tid - lrow * LG;
      int id = sIdx[k * MROWS + lrow];
      const unsigned short* src = feats + (long)(id < 0 ? 0 : id) * CIN;
      for (int c = lc0; c < CIN_PAD; c += LG)
        ldsA[lrow * CIN_PAD + c] = (id >= 0 && c < CIN) ? src[c] : (unsigned short)0;
    }
    __syncthreads();

    const unsigned short* cbuf = ldsA + buf * BUFELEMS;
#pragma unroll
    for (int tt = 0; tt < TPW; ++tt) {
      const int t = wave + tt * NWAVES;
      if (t < NTILES) {                               // wave-uniform branch
        const int col = t * 16 + mrow;
        const bool colOK = (col < COUT);
        const uint4* brow =
            (const uint4*)(Wt + ((long)k * COUT + (colOK ? col : 0)) * CIN_PAD);
#pragma unroll
        for (int c = 0; c < CHUNKS; ++c) {
          const int q0 = c * 4 + (klo >> 3);          // (c*32 + klo)/8
          union { v16bf v; uint4 q[2]; } B;
          if (colOK) { B.q[0] = brow[q0]; B.q[1] = brow[q0 + 2]; }
          else       { B.q[0] = make_uint4(0u,0u,0u,0u); B.q[1] = make_uint4(0u,0u,0u,0u); }
#pragma unroll
          for (int g = 0; g < ROWS; ++g) {            // B fragment reused
            const uint4* arow = (const uint4*)(cbuf + (g * 16 + mrow) * CIN_PAD);
            union { v16bf v; uint4 q[2]; } A;
            A.q[0] = arow[q0];
            A.q[1] = arow[q0 + 2];
            acc[tt][g] = __builtin_amdgcn_wmma_f32_16x16x32_bf16(
                false, A.v, false, B.v, (short)0, acc[tt][g], false, false);
          }
        }
      }
    }
    __syncthreads();
  }

  // Epilogue. C layout: VGPR j -> M = j + 8*(lane>=16), N = lane&15.
  const int mbase = (lane >> 4) << 3;
#pragma unroll
  for (int tt = 0; tt < TPW; ++tt) {
    const int t = wave + tt * NWAVES;
    if (t >= NTILES) continue;
    const int col = t * 16 + mrow;
    if (col >= COUT) continue;
    const float g = gamma ? gamma[col] : 1.f;
    const float b = beta ? beta[col] : 0.f;
#pragma unroll
    for (int gi = 0; gi < ROWS; ++gi) {
#pragma unroll
      for (int j = 0; j < 8; ++j) {
        const int r = row0 + gi * 16 + mbase + j;
        if (r >= n_out) continue;
        float v = acc[tt][gi][j] * g + b;
        if (resid) v += bf2f(resid[(long)r * COUT + col]);
        if (relu_flag && v < 0.f) v = 0.f;
        if (out_f32) ((float*)out)[(long)r * COUT + col] = v;
        else ((unsigned short*)out)[(long)r * COUT + col] = f2bf(v);
      }
    }
  }
}

// ---------------------------------------------------------------------------
// Host side
// ---------------------------------------------------------------------------
static inline int gsgrid(long total, int threads) {
  long b = (total + threads - 1) / threads;
  if (b > 4096) b = 4096;
  if (b < 1) b = 1;
  return (int)b;
}

template <int CIN, int COUT, bool ASYNC>
static void launch_conv_t(const unsigned short* feats, const int* nbr,
                          const unsigned short* Wt, const float* gamma,
                          const float* beta, const unsigned short* resid,
                          const unsigned short* zrow, void* out, int n_out,
                          int Kk, int relu, int outf32, hipStream_t stream) {
  constexpr int CIN_PAD = (CIN + 31) & ~31;
  constexpr int NTILES = (COUT + 15) / 16;
  constexpr int NWAVES = NTILES < 8 ? NTILES : 8;
  constexpr int MROWS = 2 * TILE_M;
  dim3 grid((n_out + MROWS - 1) / MROWS);
  dim3 block(32 * NWAVES);
  size_t shmem = (size_t)(ASYNC ? 2 : 1) * MROWS * CIN_PAD * sizeof(unsigned short);
  conv_wmma<CIN, COUT, ASYNC><<<grid, block, shmem, stream>>>(
      feats, nbr, Wt, gamma, beta, resid, zrow, out, n_out, Kk, relu, outf32);
}

extern "C" void kernel_launch(void* const* d_in, const int* in_sizes, int n_in,
                              void* d_out, int out_size, void* d_ws, size_t ws_size,
                              hipStream_t stream) {
  (void)in_sizes; (void)n_in; (void)out_size; (void)ws_size;
  const int N = 80000, ND = 10000, K = 27;
  const int C3 = 16, C2 = 64, CP = 259, CM = 80, CO = 96;

  const float* x3d    = (const float*)d_in[0];
  const float* f2d    = (const float*)d_in[1];
  const int*   nn_idx = (const int*)d_in[2];
  const int*   nbr    = (const int*)d_in[3];
  const int*   nbr_ds = (const int*)d_in[4];
  const float* W3d = (const float*)d_in[5];
  const float* g3d = (const float*)d_in[6];
  const float* b3d = (const float*)d_in[7];
  const float* Wg  = (const float*)d_in[8];
  const float* bg  = (const float*)d_in[9];
  const float* Wc  = (const float*)d_in[10];
  const float* bc  = (const float*)d_in[11];
  const float* Wp  = (const float*)d_in[12];
  const float* W2d = (const float*)d_in[13];
  const float* g2d = (const float*)d_in[14];
  const float* b2d = (const float*)d_in[15];
  const float* Wm1 = (const float*)d_in[16];
  const float* gm1 = (const float*)d_in[17];
  const float* bm1 = (const float*)d_in[18];
  const float* Wm2 = (const float*)d_in[19];
  const float* gm2 = (const float*)d_in[20];
  const float* bm2 = (const float*)d_in[21];
  const float* Wa1 = (const float*)d_in[22];
  const float* ga1 = (const float*)d_in[23];
  const float* ba1 = (const float*)d_in[24];
  const float* Wa2 = (const float*)d_in[25];
  const float* ga2 = (const float*)d_in[26];
  const float* ba2 = (const float*)d_in[27];
  const float* Wds = (const float*)d_in[28];
  const float* gds = (const float*)d_in[29];
  const float* bds = (const float*)d_in[30];

  char* ws = (char*)d_ws;
  size_t off = 0;
  auto carve = [&](size_t elems) -> unsigned short* {
    unsigned short* p = (unsigned short*)(ws + off);
    off = (off + elems * sizeof(unsigned short) + 255) & ~(size_t)255;
    return p;
  };

  // bf16 activations
  unsigned short* x3db  = carve((size_t)N * C3);
  unsigned short* f2g   = carve((size_t)N * CP);
  unsigned short* y3d   = carve((size_t)N * C3);
  unsigned short* gate  = carve((size_t)N * CP);
  unsigned short* cross = carve((size_t)N * CP);
  unsigned short* p2d   = carve((size_t)N * C2);
  unsigned short* cr2d  = carve((size_t)N * C2);
  unsigned short* x2d   = carve((size_t)N * C2);
  unsigned short* mix   = carve((size_t)N * CM);
  unsigned short* h1    = carve((size_t)N * CM);
  unsigned short* h2    = carve((size_t)N * CM);
  unsigned short* a1    = carve((size_t)N * CM);
  unsigned short* a2    = carve((size_t)N * CM);
  unsigned short* zrow  = carve(512);                 // zero source row

  // bf16 transposed / padded weights: [K][Cout][Cin_pad]
  const int P16 = 32, P64 = 64, P80 = 96, P259 = 288;
  unsigned short* Wt3d = carve((size_t)K * C3 * P16);
  unsigned short* Wtg  = carve((size_t)CP * P16);
  unsigned short* Wtc  = carve((size_t)CP * P16);
  unsigned short* Wtp  = carve((size_t)C2 * P259);
  unsigned short* Wt2d = carve((size_t)K * C2 * P64);
  unsigned short* Wtm1 = carve((size_t)K * CM * P80);
  unsigned short* Wtm2 = carve((size_t)K * CM * P80);
  unsigned short* Wta1 = carve((size_t)K * CM * P80);
  unsigned short* Wta2 = carve((size_t)K * CM * P80);
  unsigned short* Wtds = carve((size_t)K * CO * P80);

  const int EWB = 256;
  zero_u16<<<2, EWB, 0, stream>>>(zrow, 512);
  wt_prep<<<gsgrid((long)K * C3 * P16, EWB), EWB, 0, stream>>>(W3d, Wt3d, K, C3, C3, P16);
  wt_prep<<<gsgrid((long)CP * P16, EWB), EWB, 0, stream>>>(Wg, Wtg, 1, C3, CP, P16);
  wt_prep<<<gsgrid((long)CP * P16, EWB), EWB, 0, stream>>>(Wc, Wtc, 1, C3, CP, P16);
  wt_prep<<<gsgrid((long)C2 * P259, EWB), EWB, 0, stream>>>(Wp, Wtp, 1, CP, C2, P259);
  wt_prep<<<gsgrid((long)K * C2 * P64, EWB), EWB, 0, stream>>>(W2d, Wt2d, K, C2, C2, P64);
  wt_prep<<<gsgrid((long)K * CM * P80, EWB), EWB, 0, stream>>>(Wm1, Wtm1, K, CM, CM, P80);
  wt_prep<<<gsgrid((long)K * CM * P80, EWB), EWB, 0, stream>>>(Wm2, Wtm2, K, CM, CM, P80);
  wt_prep<<<gsgrid((long)K * CM * P80, EWB), EWB, 0, stream>>>(Wa1, Wta1, K, CM, CM, P80);
  wt_prep<<<gsgrid((long)K * CM * P80, EWB), EWB, 0, stream>>>(Wa2, Wta2, K, CM, CM, P80);
  wt_prep<<<gsgrid((long)K * CO * P80, EWB), EWB, 0, stream>>>(Wds, Wtds, K, CM, CO, P80);

  cvt_bf16<<<gsgrid((long)N * C3, EWB), EWB, 0, stream>>>(x3d, x3db, (long)N * C3);
  gather_cvt<<<gsgrid((long)N * CP, EWB), EWB, 0, stream>>>(f2d, nn_idx, f2g, N, CP);

  // y3d = relu(subm_conv(x3d, W3d) * g3d + b3d)
  launch_conv_t<16, 16, true>(x3db, nbr, Wt3d, g3d, b3d, nullptr, zrow, y3d, N, K, 1, 0, stream);
  // gate / cross = relu(y3d @ W + b)
  launch_conv_t<16, 259, true>(y3d, nullptr, Wtg, nullptr, bg, nullptr, zrow, gate, N, 1, 1, 0, stream);
  launch_conv_t<16, 259, true>(y3d, nullptr, Wtc, nullptr, bc, nullptr, zrow, cross, N, 1, 1, 0, stream);
  // gate *= f2g ; cross *= f2g
  ew_mul_bf16<<<gsgrid((long)N * CP, EWB), EWB, 0, stream>>>(gate, f2g, (long)N * CP);
  ew_mul_bf16<<<gsgrid((long)N * CP, EWB), EWB, 0, stream>>>(cross, f2g, (long)N * CP);
  // p2d = (gate*f2g) @ Wp ; cross2d = (cross*f2g) @ Wp   (Cin=259 -> scalar stage)
  launch_conv_t<259, 64, false>(gate, nullptr, Wtp, nullptr, nullptr, nullptr, zrow, p2d, N, 1, 0, 0, stream);
  launch_conv_t<259, 64, false>(cross, nullptr, Wtp, nullptr, nullptr, nullptr, zrow, cr2d, N, 1, 0, 0, stream);
  // x2d = relu(subm_conv(p2d, W2d) * g2d + b2d)
  launch_conv_t<64, 64, true>(p2d, nbr, Wt2d, g2d, b2d, nullptr, zrow, x2d, N, K, 1, 0, stream);
  // mix = concat(y3d, x2d + cross2d)
  concat_mix<<<gsgrid((long)N * CM, EWB), EWB, 0, stream>>>(y3d, x2d, cr2d, mix, N);
  // h = relu(conv(mix)*gm1+bm1); h = relu(conv(h)*gm2+bm2+mix)
  launch_conv_t<80, 80, true>(mix, nbr, Wtm1, gm1, bm1, nullptr, zrow, h1, N, K, 1, 0, stream);
  launch_conv_t<80, 80, true>(h1, nbr, Wtm2, gm2, bm2, mix, zrow, h2, N, K, 1, 0, stream);
  // a = relu(conv(h)*ga1+ba1); a = relu(conv(a)*ga2+ba2+h)
  launch_conv_t<80, 80, true>(h2, nbr, Wta1, ga1, ba1, nullptr, zrow, a1, N, K, 1, 0, stream);
  launch_conv_t<80, 80, true>(a1, nbr, Wta2, ga2, ba2, h2, zrow, a2, N, K, 1, 0, stream);
  // out = relu(conv_ds(a)*gds+bds)  (fp32 output)
  launch_conv_t<80, 96, true>(a2, nbr_ds, Wtds, gds, bds, nullptr, zrow, d_out, ND, K, 1, 1, stream);
}
